// CorticalColumnMLP_62182536512335
// MI455X (gfx1250) — matrix operand order repaired and checked by reading, not compiled, not run on hardware
//
#include <hip/hip_runtime.h>
#include <hip/hip_bf16.h>
#include <math.h>
#include <stdint.h>

// ---------------------------------------------------------------------------
// CorticalColumnMLP: out = gelu(x @ (w1*mask1)^T) @ (w2*mask2)^T
// M=8192, D=2048, H=8192. mask1 kills whole rows of w1 (2-of-4 per aligned
// group) and gelu(0)=0, so killed h-columns contribute nothing downstream.
// Compact to Hc=4096 surviving rows/cols -> both GEMMs at half size:
//   GEMM1: [8192x2048] x [4096x2048]^T  (gelu fused, bf16 out)
//   GEMM2: [8192x4096] x [2048x4096]^T  (fp32 out)
// GEMMs: v_wmma_f32_16x16x32_bf16, block tile 256x128, wave tile 64x64
// (16 WMMA : 16 ds_load_b128 per K-slice), BK=64, TDM (tensor_load_to_lds)
// double-buffered staging with TENSORcnt handoff; K-loop unrolled x2 so the
// buffer index is compile-time constant.
// ---------------------------------------------------------------------------

typedef __bf16 bf16_t;
typedef __attribute__((ext_vector_type(16))) __bf16 v16bf;
typedef __attribute__((ext_vector_type(8)))  __bf16 v8bf;
typedef __attribute__((ext_vector_type(8)))  float  v8f;
typedef __attribute__((ext_vector_type(4)))  unsigned int v4u;
typedef __attribute__((ext_vector_type(8)))  int v8i;
typedef __attribute__((ext_vector_type(4)))  int v4i;

static constexpr int Dd = 2048;     // n_embd
static constexpr int Hh = 4 * Dd;   // hidden = 8192
static constexpr int Hc = Hh / 2;   // kept hidden rows = 4096
static constexpr int Mm = 4 * 2048; // B*S tokens

static constexpr int BM = 256, BN = 128, BK = 64;
static constexpr int PAD = 8;          // 16B pad per 128B row -> bank spread
static constexpr int LDT = BK + PAD;   // LDS row stride (elements) = 72

#if __has_builtin(__builtin_amdgcn_tensor_load_to_lds) && \
    __has_builtin(__builtin_amdgcn_s_wait_tensorcnt)
#define USE_TDM 1
#else
#define USE_TDM 0
#endif

// ---------------- compaction / conversion passes (bandwidth-bound) ---------
__global__ void __launch_bounds__(256)
cvt_x_kernel(const float* __restrict__ x, bf16_t* __restrict__ xb, int n) {
  int i = (blockIdx.x * blockDim.x + threadIdx.x) * 4;
  if (i < n) {
    float4 v = *reinterpret_cast<const float4*>(x + i);
    bf16_t* o = xb + i;
    o[0] = (bf16_t)v.x; o[1] = (bf16_t)v.y; o[2] = (bf16_t)v.z; o[3] = (bf16_t)v.w;
  }
}

// One thread per group of 4 w1 rows: exactly 2 survive; record their indices.
__global__ void __launch_bounds__(256)
build_idx_kernel(const float* __restrict__ mask1, int* __restrict__ idx) {
  int g = blockIdx.x * blockDim.x + threadIdx.x;
  if (g < Hh / 4) {
    int base = 4 * g;
    int out0 = base, out1 = base + 1;  // fallback (mask guarantees 2 kept)
    int c = 0;
    #pragma unroll
    for (int j = 0; j < 4; ++j) {
      if (mask1[(size_t)(base + j) * Dd] != 0.0f) {
        if (c == 0) out0 = base + j; else if (c == 1) out1 = base + j;
        ++c;
      }
    }
    idx[2 * g]     = out0;
    idx[2 * g + 1] = out1;
  }
}

// w1c[r,:] = bf16(w1[idx[r],:])   (kept rows have mask==1 everywhere)
__global__ void __launch_bounds__(256)
gather_w1_kernel(const float* __restrict__ w1, const int* __restrict__ idx,
                 bf16_t* __restrict__ w1c) {
  int i = (blockIdx.x * blockDim.x + threadIdx.x) * 4;
  if (i < Hc * Dd) {
    int row = i / Dd, col = i - row * Dd;
    float4 v = *reinterpret_cast<const float4*>(w1 + (size_t)idx[row] * Dd + col);
    bf16_t* o = w1c + i;
    o[0] = (bf16_t)v.x; o[1] = (bf16_t)v.y; o[2] = (bf16_t)v.z; o[3] = (bf16_t)v.w;
  }
}

// w2c[d,j] = bf16(w2[d, idx[j]] * mask2[d, idx[j]])  (column gather)
__global__ void __launch_bounds__(256)
gather_w2_kernel(const float* __restrict__ w2, const float* __restrict__ mask2,
                 const int* __restrict__ idx, bf16_t* __restrict__ w2c) {
  int i = blockIdx.x * blockDim.x + threadIdx.x;
  if (i < Dd * Hc) {
    int d = i / Hc, j = i - d * Hc;
    size_t src = (size_t)d * Hh + idx[j];
    w2c[i] = (bf16_t)(w2[src] * mask2[src]);
  }
}

// ---------------- TDM descriptor issue (ISA 8.3/8.4) -----------------------
#if USE_TDM
// Load a [rows x 64 cols] bf16 tile (row stride K elems) into LDS at
// lds_base, padding LDS by 16B after every 128B row (-> LDT=72 elem stride).
__device__ __forceinline__ void tdm_load_tile(const bf16_t* gsrc,
                                              uint32_t lds_base, int K,
                                              int rows) {
  uint64_t ga = (uint64_t)(uintptr_t)gsrc;
  v4u g0;
  g0[0] = 1u;                                   // count=1, user mode
  g0[1] = lds_base;                             // D#.lds_addr (bytes)
  g0[2] = (uint32_t)ga;                         // global_addr[31:0]
  g0[3] = (uint32_t)((ga >> 32) & 0x01FFFFFFu)  // global_addr[56:32]
        | (2u << 30);                           // type = 2 ("image")

  const uint32_t tdim0 = (uint32_t)K;           // never OOB (tile fits)
  const uint32_t tdim1 = 1u << 20;
  const uint32_t til0 = (uint32_t)BK;           // 64 elems / row
  const uint32_t til1 = (uint32_t)rows;
  const uint64_t str0 = (uint64_t)K;            // row stride (elems)

  uint32_t w0 = (1u << 16)    // data_size = 2 bytes
              | (1u << 20)    // pad_enable
              | (4u << 22)    // pad_interval: every 32 DWORDs (=128B row)
              | (3u << 25);   // pad_amount: 4 DWORDs (=16B)
  uint32_t w1 = (tdim0 & 0xFFFFu) << 16;
  uint32_t w2 = (tdim0 >> 16) | ((tdim1 & 0xFFFFu) << 16);
  uint32_t w3 = (tdim1 >> 16) | (til0 << 16);
  uint32_t w4 = til1;                                   // tile_dim2 = 0
  uint32_t w5 = (uint32_t)str0;
  uint32_t w6 = (uint32_t)(str0 >> 32);                 // dim1_stride unused
  v8i g1 = { (int)w0, (int)w1, (int)w2, (int)w3,
             (int)w4, (int)w5, (int)w6, 0 };
  v4i z4 = { 0, 0, 0, 0 };
#if defined(__clang_major__) && __clang_major__ >= 23
  v8i z8 = { 0, 0, 0, 0, 0, 0, 0, 0 };
  __builtin_amdgcn_tensor_load_to_lds(g0, g1, z4, z4, z8, 0);
#else
  __builtin_amdgcn_tensor_load_to_lds(g0, g1, z4, z4, 0);
#endif
}
#endif  // USE_TDM

// ---------------- WMMA GEMM: Out[M,N] = A[M,K] * W[N,K]^T ------------------
__device__ __forceinline__ float gelu_exact(float x) {
  return 0.5f * x * (1.0f + erff(x * 0.70710678118654752f));
}

// 8 waves as 4(M)x2(N); wave tile 64x64 = 4x4 WMMA accumulators.
// Per K=32 slice: 8+8 ds_load_b128 feed 16 WMMAs (1:1).
__device__ __forceinline__ void compute_tile(const bf16_t* __restrict__ tA,
                                             const bf16_t* __restrict__ tB,
                                             v8f (&acc)[4][4],
                                             int wm, int wn, int r, int hi) {
  #pragma unroll
  for (int s = 0; s < 2; ++s) {
    const int kb = s * 32;
    // A fragment 16x32 bf16 (ISA 7.12.2): lanes 0-15: K=0..7,16..23;
    // lanes 16-31: K=8..15,24..31; M = lane%16.
    v16bf afrag[4];
    #pragma unroll
    for (int mi = 0; mi < 4; ++mi) {
      const int row = wm * 64 + mi * 16 + r;
      v8bf lo = *reinterpret_cast<const v8bf*>(&tA[row * LDT + kb + hi * 8]);
      v8bf hb = *reinterpret_cast<const v8bf*>(&tA[row * LDT + kb + 16 + hi * 8]);
      afrag[mi] = __builtin_shufflevector(lo, hb, 0, 1, 2, 3, 4, 5, 6, 7,
                                          8, 9, 10, 11, 12, 13, 14, 15);
    }
    // B fragment 32x16 bf16: lane = N column; lanes 0-15: K=0..7,16..23;
    // lanes 16-31: K=8..15,24..31 (same order as A).
    v16bf bfrag[4];
    #pragma unroll
    for (int ni = 0; ni < 4; ++ni) {
      const int col = wn * 64 + ni * 16 + r;
      v8bf lo = *reinterpret_cast<const v8bf*>(&tB[col * LDT + kb + hi * 8]);
      v8bf hb = *reinterpret_cast<const v8bf*>(&tB[col * LDT + kb + 16 + hi * 8]);
      bfrag[ni] = __builtin_shufflevector(lo, hb, 0, 1, 2, 3, 4, 5, 6, 7,
                                          8, 9, 10, 11, 12, 13, 14, 15);
    }
    #pragma unroll
    for (int mi = 0; mi < 4; ++mi)
      #pragma unroll
      for (int ni = 0; ni < 4; ++ni)
        acc[mi][ni] = __builtin_amdgcn_wmma_f32_16x16x32_bf16(
            false, afrag[mi], false, bfrag[ni], (short)0, acc[mi][ni],
            false, false);
  }
}

template <bool GELU_TO_BF16>
__global__ void __launch_bounds__(256)
gemm_wmma_bf16(const bf16_t* __restrict__ A, const bf16_t* __restrict__ W,
               void* __restrict__ OutP, int M, int N, int K) {
  __shared__ bf16_t sA[2][BM * LDT];   // 2 x 36 KB
  __shared__ bf16_t sB[2][BN * LDT];   // 2 x 18 KB

  const int tid  = threadIdx.x;
  const int lane = tid & 31;
  const int wave = tid >> 5;   // 0..7
  const int wm   = wave >> 1;  // 0..3 -> 64 rows each
  const int wn   = wave & 1;   // 0..1 -> 64 cols each

  const int bm = blockIdx.y * BM;
  const int bn = blockIdx.x * BN;

  const int r  = lane & 15;    // 0..15
  const int hi = lane >> 4;    // 0 or 1 (half-wave)

  v8f acc[4][4];
  #pragma unroll
  for (int mi = 0; mi < 4; ++mi)
    #pragma unroll
    for (int ni = 0; ni < 4; ++ni)
      acc[mi][ni] = v8f{};

#if USE_TDM
  // ---- TDM double-buffered staging: wave 0 drives the Tensor Data Mover ---
  const uint32_t ldsA0 = (uint32_t)(uintptr_t)&sA[0][0];
  const uint32_t ldsA1 = (uint32_t)(uintptr_t)&sA[1][0];
  const uint32_t ldsB0 = (uint32_t)(uintptr_t)&sB[0][0];
  const uint32_t ldsB1 = (uint32_t)(uintptr_t)&sB[1][0];
  const bf16_t* Abase = A + (size_t)bm * K;
  const bf16_t* Wbase = W + (size_t)bn * K;

  if (wave == 0) {
    tdm_load_tile(Abase, ldsA0, K, BM);
    tdm_load_tile(Wbase, ldsB0, K, BN);
  }
  // K/BK is even for both GEMMs (2048/64=32, 4096/64=64): unroll x2 so the
  // double-buffer selection is compile-time constant.
  for (int kt = 0; kt < K; kt += 2 * BK) {
    // --- step 0: compute buf0, prefetch into buf1 ---
    if (wave == 0) __builtin_amdgcn_s_wait_tensorcnt(0);
    __syncthreads();
    if (wave == 0) {   // kt+BK < K always holds here (K/BK even, >=2)
      tdm_load_tile(Abase + kt + BK, ldsA1, K, BM);
      tdm_load_tile(Wbase + kt + BK, ldsB1, K, BN);
    }
    compute_tile(&sA[0][0], &sB[0][0], acc, wm, wn, r, hi);
    __syncthreads();
    // --- step 1: compute buf1, prefetch into buf0 ---
    if (wave == 0) __builtin_amdgcn_s_wait_tensorcnt(0);
    __syncthreads();
    if (wave == 0 && kt + 2 * BK < K) {
      tdm_load_tile(Abase + kt + 2 * BK, ldsA0, K, BM);
      tdm_load_tile(Wbase + kt + 2 * BK, ldsB0, K, BN);
    }
    compute_tile(&sA[1][0], &sB[1][0], acc, wm, wn, r, hi);
    __syncthreads();
  }
#else
  // ---- fallback: synchronous staged copy (global -> VGPR -> LDS) ----------
  for (int kt = 0; kt < K; kt += BK) {
    #pragma unroll
    for (int e = tid; e < BM * 4; e += 256) {       // A tile: 16 elems/chunk
      int row = e >> 2, seg = (e & 3) * 16;
      const float4* g =
          reinterpret_cast<const float4*>(A + (size_t)(bm + row) * K + kt + seg);
      float4 d0 = g[0], d1 = g[1];
      float4* l = reinterpret_cast<float4*>(&sA[0][row * LDT + seg]);
      l[0] = d0; l[1] = d1;
    }
    #pragma unroll
    for (int e = tid; e < BN * 4; e += 256) {       // B tile
      int row = e >> 2, seg = (e & 3) * 16;
      const float4* g =
          reinterpret_cast<const float4*>(W + (size_t)(bn + row) * K + kt + seg);
      float4 d0 = g[0], d1 = g[1];
      float4* l = reinterpret_cast<float4*>(&sB[0][row * LDT + seg]);
      l[0] = d0; l[1] = d1;
    }
    __syncthreads();
    compute_tile(&sA[0][0], &sB[0][0], acc, wm, wn, r, hi);
    __syncthreads();
  }
#endif

  // ---- epilogue: f32 C/D layout: VGPR v -> row v + 8*hi, col = lane%16 ----
  #pragma unroll
  for (int mi = 0; mi < 4; ++mi) {
    #pragma unroll
    for (int ni = 0; ni < 4; ++ni) {
      const v8f a = acc[mi][ni];
      const int row0 = bm + wm * 64 + mi * 16 + hi * 8;
      const int col  = bn + wn * 64 + ni * 16 + r;
      if (GELU_TO_BF16) {
        bf16_t* Out = reinterpret_cast<bf16_t*>(OutP);
        #pragma unroll
        for (int v = 0; v < 8; ++v)
          Out[(size_t)(row0 + v) * N + col] = (bf16_t)gelu_exact(a[v]);
      } else {
        float* Out = reinterpret_cast<float*>(OutP);
        #pragma unroll
        for (int v = 0; v < 8; ++v)
          Out[(size_t)(row0 + v) * N + col] = a[v];
      }
    }
  }
}

// ---------------------------------------------------------------------------
extern "C" void kernel_launch(void* const* d_in, const int* in_sizes, int n_in,
                              void* d_out, int out_size, void* d_ws, size_t ws_size,
                              hipStream_t stream) {
  const float* x     = (const float*)d_in[0];  // [M, D]
  const float* w1    = (const float*)d_in[1];  // [H, D]
  const float* w2    = (const float*)d_in[2];  // [D, H]
  const float* mask1 = (const float*)d_in[3];  // [H, D]
  const float* mask2 = (const float*)d_in[4];  // [D, H]

  char* ws = (char*)d_ws;
  const size_t xb_bytes  = (size_t)Mm * Dd * sizeof(bf16_t); // 32 MB
  const size_t w1c_bytes = (size_t)Hc * Dd * sizeof(bf16_t); // 16 MB
  const size_t w2c_bytes = (size_t)Dd * Hc * sizeof(bf16_t); // 16 MB
  const size_t hb_bytes  = (size_t)Mm * Hc * sizeof(bf16_t); // 64 MB
  bf16_t* xb  = (bf16_t*)(ws);
  bf16_t* w1c = (bf16_t*)(ws + xb_bytes);
  bf16_t* w2c = (bf16_t*)(ws + xb_bytes + w1c_bytes);
  bf16_t* hb  = (bf16_t*)(ws + xb_bytes + w1c_bytes + w2c_bytes);
  int*    idx = (int*)(ws + xb_bytes + w1c_bytes + w2c_bytes + hb_bytes); // 16 KB

  // 1) structured-sparsity compaction + fp32->bf16 (pure bandwidth)
  {
    int n = Mm * Dd;
    cvt_x_kernel<<<n / 4 / 256, 256, 0, stream>>>(x, xb, n);
    build_idx_kernel<<<(Hh / 4 + 255) / 256, 256, 0, stream>>>(mask1, idx);
    n = Hc * Dd;
    gather_w1_kernel<<<n / 4 / 256, 256, 0, stream>>>(w1, idx, w1c);
    n = Dd * Hc;
    gather_w2_kernel<<<(n + 255) / 256, 256, 0, stream>>>(w2, mask2, idx, w2c);
  }

  // 2) h = gelu(x @ w1c^T) -> bf16 [M, Hc]
  {
    dim3 grid(Hc / BN, Mm / BM);
    gemm_wmma_bf16<true><<<grid, 256, 0, stream>>>(xb, w1c, (void*)hb,
                                                   Mm, Hc, Dd);
  }
  // 3) out = h @ w2c^T -> fp32 [M, D]
  {
    dim3 grid(Dd / BN, Mm / BM);
    gemm_wmma_bf16<false><<<grid, 256, 0, stream>>>(hb, w2c, d_out,
                                                    Mm, Dd, Hc);
  }
}
